// CachedMultiHeadAttention_21088289423900
// MI455X (gfx1250) — matrix operand order repaired and checked by reading, not compiled, not run on hardware
//
#include <hip/hip_runtime.h>
#include <stdint.h>

typedef __bf16 bf16;
typedef __attribute__((ext_vector_type(16))) __bf16 v16bf;
typedef __attribute__((ext_vector_type(8)))  __bf16 v8bf;
typedef __attribute__((ext_vector_type(8)))  float  v8f;
typedef __attribute__((ext_vector_type(4)))  float  v4f;

#define BATCH 2
#define SEQ   2048
#define DIN   2048
#define DOUT  2048
#define NH    16
#define HD    128
#define QK_SCALE 0.08838834764831845f   // 1/sqrt(128)

static __device__ __forceinline__ bf16 f2bf(float f) {
  uint32_t u = __builtin_bit_cast(uint32_t, f);
  uint32_t r = u + 0x7FFFu + ((u >> 16) & 1u);   // round-to-nearest-even
  uint16_t h = (uint16_t)(r >> 16);
  return __builtin_bit_cast(bf16, h);
}

static __device__ __forceinline__ v16bf cat8(v8bf lo, v8bf hi) {
  return __builtin_shufflevector(lo, hi, 0,1,2,3,4,5,6,7,8,9,10,11,12,13,14,15);
}

static __device__ __forceinline__ v8f wmma_bf16(v16bf a, v16bf b, v8f c) {
  // D = A(16x32 bf16) * B(32x16 bf16) + C(16x16 f32)
  return __builtin_amdgcn_wmma_f32_16x16x32_bf16(false, a, false, b, (short)0, c, false, false);
}

// ---------------------------------------------------------------------------
// fp32 -> bf16 cast, 8 elements / thread (n must be a multiple of 8)
// ---------------------------------------------------------------------------
__global__ void cast_f32_to_bf16(const float* __restrict__ src, bf16* __restrict__ dst, int n) {
  int i = (blockIdx.x * blockDim.x + threadIdx.x) * 8;
  if (i < n) {
    v4f f0 = *(const v4f*)(src + i);
    v4f f1 = *(const v4f*)(src + i + 4);
    v8bf o;
    o[0] = f2bf(f0[0]); o[1] = f2bf(f0[1]); o[2] = f2bf(f0[2]); o[3] = f2bf(f0[3]);
    o[4] = f2bf(f1[0]); o[5] = f2bf(f1[1]); o[6] = f2bf(f1[2]); o[7] = f2bf(f1[3]);
    *(v8bf*)(dst + i) = o;
  }
}

// ---------------------------------------------------------------------------
// QKV GEMM: C[4096 x 6144] = X[4096 x 2048] * Wqkv[6144 x 2048]^T
// Per wave: 64(M) x 64(N) tile; 8 waves -> 128 x 256 block tile.
// Epilogue scatters to Q[B,H,S,Dh] (scaled), K[B,H,S,Dh], Vt[B,H,Dh,S].
// ---------------------------------------------------------------------------
__global__ __launch_bounds__(256) void qkv_gemm(
    const bf16* __restrict__ A, const bf16* __restrict__ W,
    bf16* __restrict__ Qb, bf16* __restrict__ Kb, bf16* __restrict__ Vt)
{
  const int lane = threadIdx.x & 31;
  const int w    = threadIdx.x >> 5;
  const int wm = w >> 2, wn = w & 3;
  const int ll = lane & 15, lh = lane >> 4;
  const int Mb = blockIdx.y * 128 + wm * 64;
  const int Nb = blockIdx.x * 256 + wn * 64;
  const int K  = DIN;

  // A-operand: lane ll = row, lh selects K range 0..15 vs 16..31
  const bf16* ap[4];
#pragma unroll
  for (int i = 0; i < 4; i++) ap[i] = A + (size_t)(Mb + i * 16 + ll) * K + lh * 16;
  // B-operand: lane ll = column (weight row), lh selects K range 0..15 vs 16..31
  const bf16* bp[4];
#pragma unroll
  for (int j = 0; j < 4; j++) bp[j] = W + (size_t)(Nb + j * 16 + ll) * K + lh * 16;

  const v8f vz = {0.f,0.f,0.f,0.f,0.f,0.f,0.f,0.f};
  v8f acc[4][4];
#pragma unroll
  for (int i = 0; i < 4; i++)
#pragma unroll
    for (int j = 0; j < 4; j++) acc[i][j] = vz;

  for (int k = 0; k < K; k += 32) {
    __builtin_prefetch(bp[0] + k + 512, 0, 3);
    __builtin_prefetch(bp[2] + k + 512, 0, 3);
    __builtin_prefetch(ap[0] + k + 512, 0, 3);
    __builtin_prefetch(ap[2] + k + 512, 0, 3);
    v16bf av[4], bv[4];
#pragma unroll
    for (int i = 0; i < 4; i++)
      av[i] = cat8(*(const v8bf*)(ap[i] + k), *(const v8bf*)(ap[i] + k + 8));
#pragma unroll
    for (int j = 0; j < 4; j++)
      bv[j] = *(const v16bf*)(bp[j] + k);
#pragma unroll
    for (int i = 0; i < 4; i++)
#pragma unroll
      for (int j = 0; j < 4; j++)
        acc[i][j] = wmma_bf16(av[i], bv[j], acc[i][j]);
  }

#pragma unroll
  for (int j = 0; j < 4; j++) {
    int col   = Nb + j * 16 + ll;
    int which = col >> 11;          // 0=Q, 1=K, 2=V
    int h     = (col >> 7) & (NH - 1);
    int d     = col & (HD - 1);
#pragma unroll
    for (int i = 0; i < 4; i++) {
#pragma unroll
      for (int r = 0; r < 8; r++) {
        int row = Mb + i * 16 + r + 8 * lh;   // C layout: M = r + 8*(lane>>4)
        int bb  = row >> 11;
        int s   = row & (SEQ - 1);
        float v = acc[i][j][r];
        if (which == 0) {
          Qb[((size_t)(bb * NH + h) * SEQ + s) * HD + d] = f2bf(v * QK_SCALE);
        } else if (which == 1) {
          Kb[((size_t)(bb * NH + h) * SEQ + s) * HD + d] = f2bf(v);
        } else {
          Vt[((size_t)(bb * NH + h) * HD + d) * SEQ + s] = f2bf(v);   // transposed V
        }
      }
    }
  }
}

// ---------------------------------------------------------------------------
// Causal flash attention, 4 cooperative waves per block.
// Block handles 64 q rows (4 x 16-row tiles); K/V staged in LDS once per
// 64-key block and consumed by all 4 waves. Blocks past a row's causal bound
// are numerical no-ops (f=1, p=0), so the trip count is uniform.
// ---------------------------------------------------------------------------
__global__ __launch_bounds__(128) void flash_attn(
    const bf16* __restrict__ Qb, const bf16* __restrict__ Kb,
    const bf16* __restrict__ Vt, bf16* __restrict__ Ob)
{
  const int tid  = threadIdx.x;
  const int lane = tid & 31;
  const int w    = tid >> 5;                  // wave 0..3
  const int ll = lane & 15, lh = lane >> 4;
  const int id = blockIdx.x;
  const int g  = id & (SEQ / 64 - 1);         // 64-row q group
  const int h  = (id >> 5) & (NH - 1);
  const int b  = id >> 9;
  const int qt = g * 4 + w;                   // this wave's 16-row q tile

  const bf16* Q  = Qb + (size_t)(b * NH + h) * SEQ * HD;
  const bf16* Kp = Kb + (size_t)(b * NH + h) * SEQ * HD;
  const bf16* Vp = Vt + (size_t)(b * NH + h) * HD * SEQ;

  __shared__ __align__(32) bf16 Klds[64 * 128];     // 16 KB: 64 keys x 128 dh
  __shared__ __align__(32) bf16 Vlds[128 * 64];     // 16 KB: 128 dh x 64 keys
  __shared__ __align__(32) bf16 Pl[4][16][64];      //  8 KB: per-wave P staging

  // Q as 4 WMMA-A operands (Dh = 128 = 4 * K32)
  v16bf qa[4];
  {
    const bf16* qrow = Q + (size_t)(qt * 16 + ll) * HD + lh * 16;
#pragma unroll
    for (int c = 0; c < 4; c++)
      qa[c] = cat8(*(const v8bf*)(qrow + c * 32), *(const v8bf*)(qrow + c * 32 + 8));
  }

  const v8f vz = {0.f,0.f,0.f,0.f,0.f,0.f,0.f,0.f};
  float m_r[8], l_r[8];
  v8f acc[8];
#pragma unroll
  for (int r = 0; r < 8; r++) { m_r[r] = -1e30f; l_r[r] = 0.0f; }
#pragma unroll
  for (int d8 = 0; d8 < 8; d8++) acc[d8] = vz;

  const int kendu = g * 64 + 64;              // uniform trip count for the block
  for (int kb = 0; kb < kendu; kb += 64) {
    // --- cooperative staging of K (row-major) and Vt (dh-major) blocks ---
#pragma unroll
    for (int it = 0; it < 4; it++) {
      int idx = tid + it * 128;               // 0..511, 32-byte chunks
      int krow = idx >> 3, kseg = idx & 7;
      *(v16bf*)&Klds[krow * 128 + kseg * 16] =
          *(const v16bf*)&Kp[(size_t)(kb + krow) * HD + kseg * 16];
      int vrow = idx >> 2, vseg = idx & 3;
      *(v16bf*)&Vlds[vrow * 64 + vseg * 16] =
          *(const v16bf*)&Vp[(size_t)vrow * SEQ + kb + vseg * 16];
    }
    __syncthreads();

    // --- scores: four 16x16 tiles over the 64-key block, fp32 accumulate ---
    v8f s[4];
#pragma unroll
    for (int t = 0; t < 4; t++) {
      s[t] = vz;
#pragma unroll
      for (int c = 0; c < 4; c++)
        s[t] = wmma_bf16(qa[c], *(const v16bf*)&Klds[(t * 16 + ll) * 128 + c * 32 + lh * 16], s[t]);
    }

    // --- causal mask + online softmax (rows live in 16-lane halves) ---
    float p[4][8];
#pragma unroll
    for (int r = 0; r < 8; r++) {
      int qrow = qt * 16 + r + 8 * lh;
      float xs[4];
#pragma unroll
      for (int t = 0; t < 4; t++)
        xs[t] = ((kb + t * 16 + ll) <= qrow) ? s[t][r] : -1e30f;
      float mx = fmaxf(fmaxf(xs[0], xs[1]), fmaxf(xs[2], xs[3]));
#pragma unroll
      for (int off = 8; off >= 1; off >>= 1)
        mx = fmaxf(mx, __shfl_xor(mx, off, 32));
      float mn = fmaxf(m_r[r], mx);
      float f  = __expf(m_r[r] - mn);
      m_r[r] = mn;
      float rs = 0.0f;
#pragma unroll
      for (int t = 0; t < 4; t++) { p[t][r] = __expf(xs[t] - mn); rs += p[t][r]; }
#pragma unroll
      for (int off = 8; off >= 1; off >>= 1)
        rs += __shfl_xor(rs, off, 32);
      l_r[r] = l_r[r] * f + rs;
#pragma unroll
      for (int d8 = 0; d8 < 8; d8++) acc[d8][r] *= f;
    }

    // --- P: C-layout -> A-layout via per-wave LDS slice ---
#pragma unroll
    for (int r = 0; r < 8; r++) {
      int row = r + 8 * lh;
#pragma unroll
      for (int t = 0; t < 4; t++)
        Pl[w][row][t * 16 + ll] = f2bf(p[t][r]);
    }
    __syncthreads();
    v16bf pa[2];
#pragma unroll
    for (int c2 = 0; c2 < 2; c2++) {
      const bf16* pr = &Pl[w][ll][c2 * 32 + lh * 16];
      pa[c2] = cat8(*(const v8bf*)pr, *(const v8bf*)(pr + 8));
    }

    // --- acc += P(16x64) @ V(64x128) from LDS (transposed V) ---
#pragma unroll
    for (int c2 = 0; c2 < 2; c2++) {
#pragma unroll
      for (int d8 = 0; d8 < 8; d8++) {
        v16bf vb = *(const v16bf*)&Vlds[(d8 * 16 + ll) * 64 + c2 * 32 + lh * 16];
        acc[d8] = wmma_bf16(pa[c2], vb, acc[d8]);
      }
    }
    __syncthreads();   // protect Klds/Vlds before next staging pass
  }

  // --- normalize and store O in [B, S, H*Dh] bf16 ---
#pragma unroll
  for (int r = 0; r < 8; r++) {
    float inv = 1.0f / l_r[r];
    int q = qt * 16 + r + 8 * lh;
#pragma unroll
    for (int d8 = 0; d8 < 8; d8++) {
      int col = h * HD + d8 * 16 + ll;
      Ob[(size_t)(b * SEQ + q) * DOUT + col] = f2bf(acc[d8][r] * inv);
    }
  }
}

// ---------------------------------------------------------------------------
// Output projection: out[4096 x 2048] = O[4096 x 2048] * Wp[2048 x 2048]^T + bias
// ---------------------------------------------------------------------------
__global__ __launch_bounds__(256) void proj_gemm(
    const bf16* __restrict__ A, const bf16* __restrict__ W,
    const float* __restrict__ bias, float* __restrict__ out)
{
  const int lane = threadIdx.x & 31;
  const int w    = threadIdx.x >> 5;
  const int wm = w >> 2, wn = w & 3;
  const int ll = lane & 15, lh = lane >> 4;
  const int Mb = blockIdx.y * 128 + wm * 64;
  const int Nb = blockIdx.x * 256 + wn * 64;
  const int K  = DOUT;

  const bf16* ap[4];
#pragma unroll
  for (int i = 0; i < 4; i++) ap[i] = A + (size_t)(Mb + i * 16 + ll) * K + lh * 16;
  const bf16* bp[4];
#pragma unroll
  for (int j = 0; j < 4; j++) bp[j] = W + (size_t)(Nb + j * 16 + ll) * K + lh * 16;

  const v8f vz = {0.f,0.f,0.f,0.f,0.f,0.f,0.f,0.f};
  v8f acc[4][4];
#pragma unroll
  for (int i = 0; i < 4; i++)
#pragma unroll
    for (int j = 0; j < 4; j++) acc[i][j] = vz;

  for (int k = 0; k < K; k += 32) {
    __builtin_prefetch(bp[0] + k + 512, 0, 3);
    __builtin_prefetch(bp[2] + k + 512, 0, 3);
    __builtin_prefetch(ap[0] + k + 512, 0, 3);
    __builtin_prefetch(ap[2] + k + 512, 0, 3);
    v16bf av[4], bv[4];
#pragma unroll
    for (int i = 0; i < 4; i++)
      av[i] = cat8(*(const v8bf*)(ap[i] + k), *(const v8bf*)(ap[i] + k + 8));
#pragma unroll
    for (int j = 0; j < 4; j++)
      bv[j] = *(const v16bf*)(bp[j] + k);
#pragma unroll
    for (int i = 0; i < 4; i++)
#pragma unroll
      for (int j = 0; j < 4; j++)
        acc[i][j] = wmma_bf16(av[i], bv[j], acc[i][j]);
  }

#pragma unroll
  for (int j = 0; j < 4; j++) {
    int col = Nb + j * 16 + ll;
    float bvx = bias[col];
#pragma unroll
    for (int i = 0; i < 4; i++) {
#pragma unroll
      for (int r = 0; r < 8; r++) {
        int row = Mb + i * 16 + r + 8 * lh;
        out[(size_t)row * DOUT + col] = acc[i][j][r] + bvx;
      }
    }
  }
}

// ---------------------------------------------------------------------------
extern "C" void kernel_launch(void* const* d_in, const int* in_sizes, int n_in,
                              void* d_out, int out_size, void* d_ws, size_t ws_size,
                              hipStream_t stream)
{
  (void)in_sizes; (void)n_in; (void)out_size; (void)ws_size;
  const float* x      = (const float*)d_in[0];
  const float* w_qkv  = (const float*)d_in[1];
  const float* w_proj = (const float*)d_in[2];
  const float* b_proj = (const float*)d_in[3];
  float* out = (float*)d_out;

  char* ws = (char*)d_ws;
  size_t off = 0;
  bf16* xb     = (bf16*)(ws + off); off += (size_t)BATCH * SEQ * DIN * 2;      // 16 MB
  bf16* wqkvb  = (bf16*)(ws + off); off += (size_t)3 * DOUT * DIN * 2;        // 24 MB
  bf16* wprojb = (bf16*)(ws + off); off += (size_t)DOUT * DOUT * 2;           //  8 MB
  bf16* Qb     = (bf16*)(ws + off); off += (size_t)BATCH * NH * SEQ * HD * 2; // 16 MB
  bf16* Kb     = (bf16*)(ws + off); off += (size_t)BATCH * NH * SEQ * HD * 2; // 16 MB
  bf16* Vt     = (bf16*)(ws + off); off += (size_t)BATCH * NH * HD * SEQ * 2; // 16 MB
  bf16* Ob     = (bf16*)(ws + off); off += (size_t)BATCH * SEQ * DOUT * 2;    // 16 MB

  const int n1 = BATCH * SEQ * DIN;
  const int n2 = 3 * DOUT * DIN;
  const int n3 = DOUT * DOUT;
  cast_f32_to_bf16<<<(n1 / 8 + 255) / 256, 256, 0, stream>>>(x,      xb,     n1);
  cast_f32_to_bf16<<<(n2 / 8 + 255) / 256, 256, 0, stream>>>(w_qkv,  wqkvb,  n2);
  cast_f32_to_bf16<<<(n3 / 8 + 255) / 256, 256, 0, stream>>>(w_proj, wprojb, n3);

  qkv_gemm <<<dim3(3 * DOUT / 256, BATCH * SEQ / 128), 256, 0, stream>>>(xb, wqkvb, Qb, Kb, Vt);
  flash_attn<<<BATCH * NH * (SEQ / 64), 128, 0, stream>>>(Qb, Kb, Vt, Ob);
  proj_gemm<<<dim3(DOUT / 256, BATCH * SEQ / 128), 256, 0, stream>>>(Ob, wprojb, b_proj, out);
}
